// IDA_pare_46986942218480
// MI455X (gfx1250) — matrix operand order repaired and checked by reading, not compiled, not run on hardware
//
#include <hip/hip_runtime.h>
#include <hip/hip_bf16.h>

// ---------------------------------------------------------------------------
// MI455X / gfx1250 IDA-Up node: DCN+BN+ReLU -> x2 bilinear upsample+residual
// -> DCN+BN+ReLU. All GEMMs (einsum M=16384/65536, N=128, K=2304/1152, and the
// two 3x3 offset convs with N=27 padded to 32) run on v_wmma_f32_16x16x32_bf16
// with fp32 accumulation. A-tiles are built on the fly into a double-buffered
// LDS tile (one barrier per K-chunk, build of chunk i+1 overlaps WMMAs of
// chunk i). Sampling geometry is register-resident per thread (its pixel is
// loop-invariant) and recomputed only when the 3x3 tap changes. Gathers use
// uniform-base + 32-bit per-lane offsets (SADDR/GVS form). ~52MB workspace,
// fully resident in the 192MB L2.
// ---------------------------------------------------------------------------

typedef __attribute__((ext_vector_type(16))) __bf16 v16bf;
typedef __attribute__((ext_vector_type(8)))  __bf16 v8bf;
typedef __attribute__((ext_vector_type(8)))  float  v8f;

// ---------------------------------------------------------------------------
// Repack conv weights (N, Cin, 3, 3) fp32 -> (Npad, 9*Cin) bf16 row-major,
// K index = tap*Cin + c (each 32-wide K chunk has a fixed tap). Rows n >= N
// are zero so the N=27 convs run as full 32-column WMMAs.
// ---------------------------------------------------------------------------
__global__ void repack_w_kernel(const float* __restrict__ src,
                                __bf16* __restrict__ dst,
                                int N, int Npad, int Cin) {
  const int K = 9 * Cin;
  const size_t total = (size_t)Npad * K;
  for (size_t i = (size_t)blockIdx.x * blockDim.x + threadIdx.x; i < total;
       i += (size_t)gridDim.x * blockDim.x) {
    const int n  = (int)(i / K);
    const int kk = (int)(i % K);
    const int k9 = kk / Cin;
    const int c  = kk % Cin;
    float v = 0.0f;
    if (n < N) v = src[((size_t)n * Cin + c) * 9 + k9];
    dst[i] = (__bf16)v;
  }
}

struct Geom { int o0, o1, o2, o3; float w0, w1, w2, w3; };

// ---------------------------------------------------------------------------
// Fused im2col / deformable-sampling implicit GEMM with WMMA.
//   BLOCK = 256 threads = 8 waves, MT = 64 pixels per block.
//   Wave w owns N-tile (w % NTILES) and M-subtiles ms = w/NTILES stepping by
//   8/NTILES (main GEMM: 8 N-tiles x 4 acc; offset conv: 2 N-tiles x 1 acc).
//   CPT = Cin/32 channel-chunks per tap (template -> shift/mask bookkeeping).
//   DEFORM=false -> regular 3x3 im2col; DEFORM=true -> bilinear * sigmoid
//   mask, 4-tap fma gather. LDS A tile is double buffered and XOR-swizzled at
//   16B granularity (conflict-free-ish ds_store_b16, aligned ds_load_b128).
// ---------------------------------------------------------------------------
template <bool DEFORM, int NTILES, int MSUB, int CPT>
__global__ void dcn_gemm_kernel(const float* __restrict__ x,    // [B,Cin,H,W]
                                const float* __restrict__ om,   // [B,27,H,W]
                                const __bf16* __restrict__ wbt, // [Npad,9*Cin]
                                const float* __restrict__ bias,
                                const float* __restrict__ gamma,
                                const float* __restrict__ beta,
                                const float* __restrict__ mean,
                                const float* __restrict__ var,
                                float* __restrict__ out,        // [B,Ovalid,H,W]
                                int H, int W, int Ovalid,
                                int hwShift, int wShift) {
  constexpr int BLOCK  = 256;
  constexpr int MT     = 16 * MSUB;             // pixels per block (64)
  constexpr int MSPW   = (NTILES * MSUB) / 8;   // M-subtiles per wave
  constexpr int MSSTEP = 8 / NTILES;
  constexpr int Cin    = CPT * 32;
  constexpr int K      = 9 * Cin;
  constexpr int NCHUNK = 9 * CPT;
  constexpr int clStep = BLOCK / MT;            // channel stride per thread
  constexpr int perThd = 32 / clStep;           // A elements built per chunk

  const int HW = H * W;
  const int m0 = blockIdx.x * MT;
  const int tid  = threadIdx.x;
  const int lane = tid & 31;
  const int wave = tid >> 5;

  __shared__ __bf16 aLds[2][MT * 32];

  // ---- loop-invariant per-thread pixel (A-build role) --------------------
  const int p   = tid & (MT - 1);               // my pixel within the tile
  const int m   = m0 + p;
  const int b   = m >> hwShift;
  const int hw  = m & (HW - 1);
  const int hh  = hw >> wShift;
  const int ww  = hw & (W - 1);
  const int xbase  = b * Cin * HW;              // 32-bit plane base
  const int swz    = (p + (p >> 2)) & 3;        // 16B-group XOR swizzle
  const int aRowB  = p * 32;                    // halfs
  const int clBase = tid / MT;
  const int stepHW = clStep * HW;

  // ---- WMMA fragment addressing (consumer role) --------------------------
  const int ntile    = wave % NTILES;
  const int msStart  = wave / NTILES;
  const int col      = ntile * 16 + (lane & 15);
  const int kHalfOff = (lane < 16) ? 0 : 8;     // B: K {0..7,16..23} / {8..15,24..31}
  const int fRowSel  = lane & 15;
  const int gA       = (lane < 16) ? 0 : 1;     // A: groups {0,2} / {1,3}

  // ---- per-tap sampling geometry, register resident ----------------------
  auto geometry = [&](int k9) -> Geom {
    Geom g;
    const int ky = k9 / 3 - 1;
    const int kx = k9 - (k9 / 3) * 3 - 1;
    if constexpr (!DEFORM) {
      const int yi = hh + ky, xi = ww + kx;
      const bool inb = (yi >= 0) & (yi < H) & (xi >= 0) & (xi < W);
      g.o0 = inb ? (yi * W + xi) : 0;
      g.w0 = inb ? 1.0f : 0.0f;
      g.o1 = g.o2 = g.o3 = 0;
      g.w1 = g.w2 = g.w3 = 0.0f;
    } else {
      const int ombase = b * 27 * HW + hw;
      const float of1 = om[ombase + k9 * HW];
      const float of2 = om[ombase + (9 + k9) * HW];
      float mk        = om[ombase + (18 + k9) * HW];
      mk = 1.0f / (1.0f + __expf(-mk));                 // sigmoid mask
      const float py  = of1 + (float)(hh + ky);
      const float px  = of2 + (float)(ww + kx);
      const float y0f = floorf(py), x0f = floorf(px);
      const int   y0  = (int)y0f,   x0 = (int)x0f;
      const float wy  = py - y0f,   wx = px - x0f;
      const bool vy0 = (y0 >= 0) & (y0 < H);
      const bool vy1 = (y0 + 1 >= 0) & (y0 + 1 < H);
      const bool vx0 = (x0 >= 0) & (x0 < W);
      const bool vx1 = (x0 + 1 >= 0) & (x0 + 1 < W);
      const int yc0 = min(max(y0, 0), H - 1);
      const int yc1 = min(max(y0 + 1, 0), H - 1);
      const int xc0 = min(max(x0, 0), W - 1);
      const int xc1 = min(max(x0 + 1, 0), W - 1);
      g.o0 = yc0 * W + xc0; g.w0 = (vy0 & vx0) ? (1.f - wy) * (1.f - wx) * mk : 0.f;
      g.o1 = yc0 * W + xc1; g.w1 = (vy0 & vx1) ? (1.f - wy) * wx * mk : 0.f;
      g.o2 = yc1 * W + xc0; g.w2 = (vy1 & vx0) ? wy * (1.f - wx) * mk : 0.f;
      g.o3 = yc1 * W + xc1; g.w3 = (vy1 & vx1) ? wy * wx * mk : 0.f;
    }
    return g;
  };

  // ---- build my slice of one MTx32 bf16 A tile (32-bit offsets) ----------
  auto build = [&](__bf16* buf, const Geom& g, int ccInTap) {
    int idx = xbase + (ccInTap * 32 + clBase) * HW;
#pragma unroll
    for (int j = 0; j < perThd; ++j) {
      const int cl = clBase + j * clStep;
      float val;
      if constexpr (!DEFORM) {
        val = g.w0 * x[idx + g.o0];
      } else {
        val = g.w0 * x[idx + g.o0] + g.w1 * x[idx + g.o1] +
              g.w2 * x[idx + g.o2] + g.w3 * x[idx + g.o3];
      }
      const int grp = cl >> 3;
      buf[aRowB + ((grp ^ swz) << 3) + (cl & 7)] = (__bf16)val;
      idx += stepHW;
    }
  };

  v8f acc[MSPW];
#pragma unroll
  for (int i = 0; i < MSPW; ++i)
#pragma unroll
    for (int j = 0; j < 8; ++j) acc[i][j] = 0.0f;

  // ---- software-pipelined main loop: one barrier per K-chunk -------------
  Geom g = geometry(0);
  build(aLds[0], g, 0);
  __syncthreads();

  for (int chk = 0; chk < NCHUNK; ++chk) {
    const int cur = chk & 1;

    if (chk + 1 < NCHUNK) {                     // build next chunk
      const int nk = chk + 1;
      if ((nk & (CPT - 1)) == 0) g = geometry(nk / CPT);
      build(aLds[cur ^ 1], g, nk & (CPT - 1));
    }

    // B fragment (shared across this wave's M-subtiles)
    const __bf16* __restrict__ brow =
        wbt + (size_t)col * K + chk * 32 + kHalfOff;
    const v8bf bLo = *(const v8bf*)(brow);
    const v8bf bHi = *(const v8bf*)(brow + 16);
    v16bf bfrag;
#pragma unroll
    for (int i = 0; i < 8; ++i) { bfrag[i] = bLo[i]; bfrag[8 + i] = bHi[i]; }
    if (chk + 1 < NCHUNK) __builtin_prefetch(brow + 32, 0, 0);

    // A fragments + WMMA accumulate
#pragma unroll
    for (int i = 0; i < MSPW; ++i) {
      const int ms = msStart + i * MSSTEP;
      const int pr = ms * 16 + fRowSel;
      const int s  = (pr + (pr >> 2)) & 3;
      const v8bf aLo = *(const v8bf*)(&aLds[cur][pr * 32 + ((gA ^ s) << 3)]);
      const v8bf aHi =
          *(const v8bf*)(&aLds[cur][pr * 32 + (((gA + 2) ^ s) << 3)]);
      v16bf afrag;
#pragma unroll
      for (int q = 0; q < 8; ++q) { afrag[q] = aLo[q]; afrag[8 + q] = aHi[q]; }
      acc[i] = __builtin_amdgcn_wmma_f32_16x16x32_bf16(
          false, afrag, false, bfrag, (short)0, acc[i], false, false);
    }
    __syncthreads();
  }

  // ---- epilogue: bias (+BN+ReLU), scatter to NCHW ------------------------
  if (col < Ovalid) {
    float sc = 1.0f, sh = bias[col];
    if (gamma != nullptr) {
      const float inv = gamma[col] * rsqrtf(var[col] + 1e-5f);
      sc = inv;
      sh = (sh - mean[col]) * inv + beta[col];
    }
#pragma unroll
    for (int i = 0; i < MSPW; ++i) {
      const int ms = msStart + i * MSSTEP;
#pragma unroll
      for (int r = 0; r < 8; ++r) {
        const int M   = ms * 16 + ((lane < 16) ? r : (8 + r));
        const int mm  = m0 + M;
        const int bb  = mm >> hwShift;
        const int phw = mm & (HW - 1);
        float v = acc[i][r] * sc + sh;
        if (gamma != nullptr) v = fmaxf(v, 0.0f);
        out[((size_t)bb * Ovalid + col) * HW + phw] = v;
      }
    }
  }
}

// ---------------------------------------------------------------------------
// Depthwise transposed-conv bilinear x2 upsample (4x4 kernel, lhs_dilation=2,
// pad=2; reference pre-flips the kernel, conv flips again -> index
// up_w[c][3-ky][3-kx]); fused with the pre_x residual add. Exactly 2x2 taps
// contribute per output pixel (parity match with the dilated grid).
// ---------------------------------------------------------------------------
__global__ void upsample_add_kernel(const float* __restrict__ h,
                                    const float* __restrict__ pre,
                                    const float* __restrict__ up_w,
                                    float* __restrict__ z,
                                    int B, int C, int H, int W) {
  const int H2 = 2 * H, W2 = 2 * W;
  const size_t total = (size_t)B * C * H2 * W2;
  for (size_t i = (size_t)blockIdx.x * blockDim.x + threadIdx.x; i < total;
       i += (size_t)gridDim.x * blockDim.x) {
    const int xo = (int)(i % W2);
    size_t t = i / W2;
    const int yo = (int)(t % H2);
    t /= H2;
    const int c = (int)(t % C);
    const int b = (int)(t / C);
    const float* __restrict__ hp = h + ((size_t)b * C + c) * H * W;
    const float* __restrict__ wp = up_w + (size_t)c * 16;
    float accv = 0.0f;
#pragma unroll
    for (int dy = 0; dy < 2; ++dy) {
      const int ky = (yo & 1) + 2 * dy;
      const int q  = yo - 2 + ky;               // dilated-input row (even)
      if (q < 0 || q > 2 * H - 2) continue;
      const int yi = q >> 1;
#pragma unroll
      for (int dx = 0; dx < 2; ++dx) {
        const int kx = (xo & 1) + 2 * dx;
        const int r  = xo - 2 + kx;
        if (r < 0 || r > 2 * W - 2) continue;
        const int xi = r >> 1;
        accv += wp[(3 - ky) * 4 + (3 - kx)] * hp[yi * W + xi];
      }
    }
    z[i] = accv + pre[i];
  }
}

// ---------------------------------------------------------------------------
// Host-side orchestration.
// ---------------------------------------------------------------------------
extern "C" void kernel_launch(void* const* d_in, const int* in_sizes, int n_in,
                              void* d_out, int out_size, void* d_ws,
                              size_t ws_size, hipStream_t stream) {
  (void)in_sizes; (void)n_in; (void)out_size; (void)ws_size;

  const float* x          = (const float*)d_in[0];   // [2,256,64,128]
  const float* pre_x      = (const float*)d_in[1];   // [2,128,128,256]
  const float* proj_w_om  = (const float*)d_in[2];   // [27,256,3,3]
  const float* proj_b_om  = (const float*)d_in[3];
  const float* proj_w     = (const float*)d_in[4];   // [128,256,3,3]
  const float* proj_b     = (const float*)d_in[5];
  const float* proj_gamma = (const float*)d_in[6];
  const float* proj_beta  = (const float*)d_in[7];
  const float* proj_mean  = (const float*)d_in[8];
  const float* proj_var   = (const float*)d_in[9];
  const float* node_w_om  = (const float*)d_in[10];  // [27,128,3,3]
  const float* node_b_om  = (const float*)d_in[11];
  const float* node_w     = (const float*)d_in[12];  // [128,128,3,3]
  const float* node_b     = (const float*)d_in[13];
  const float* node_gamma = (const float*)d_in[14];
  const float* node_beta  = (const float*)d_in[15];
  const float* node_mean  = (const float*)d_in[16];
  const float* node_var   = (const float*)d_in[17];
  const float* up_w       = (const float*)d_in[18];  // [128,1,4,4]

  const int B = 2, C = 256, O = 128, H = 64, W = 128;
  const int H2 = 2 * H, W2 = 2 * W;
  const int HW = H * W, HW2 = H2 * W2;

  // ---- workspace layout (256B aligned slabs), ~52 MB total ----
  char* ws = (char*)d_ws;
  size_t off = 0;
  auto slab = [&](size_t bytes) {
    char* p = ws + off;
    off += (bytes + 255) & ~(size_t)255;
    return p;
  };
  __bf16* WB1  = (__bf16*)slab((size_t)32  * 9 * C * 2);  // proj offset conv W
  __bf16* WB2  = (__bf16*)slab((size_t)128 * 9 * C * 2);  // proj dcn W
  __bf16* WB3  = (__bf16*)slab((size_t)32  * 9 * O * 2);  // node offset conv W
  __bf16* WB4  = (__bf16*)slab((size_t)128 * 9 * O * 2);  // node dcn W
  float*  om1  = (float*)slab((size_t)B * 27 * HW  * 4);  // proj offsets/mask
  float*  om2  = (float*)slab((size_t)B * 27 * HW2 * 4);  // node offsets/mask
  float*  hbuf = (float*)slab((size_t)B * O  * HW  * 4);  // proj DCN out
  float*  zbuf = (float*)slab((size_t)B * O  * HW2 * 4);  // upsample + pre_x

  // ---- weight repacks (fp32 -> bf16, K = tap*Cin + c) ----
  repack_w_kernel<<<256, 256, 0, stream>>>(proj_w_om, WB1, 27, 32, C);
  repack_w_kernel<<<512, 256, 0, stream>>>(proj_w,    WB2, 128, 128, C);
  repack_w_kernel<<<128, 256, 0, stream>>>(node_w_om, WB3, 27, 32, O);
  repack_w_kernel<<<256, 256, 0, stream>>>(node_w,    WB4, 128, 128, O);

  // ---- proj block (HW = 8192 -> shift 13, W = 128 -> shift 7) ----
  const int M1 = B * HW;                              // 16384 pixels
  dcn_gemm_kernel<false, 2, 4, 8><<<M1 / 64, 256, 0, stream>>>(
      x, nullptr, WB1, proj_b_om, nullptr, nullptr, nullptr, nullptr,
      om1, H, W, 27, 13, 7);
  dcn_gemm_kernel<true, 8, 4, 8><<<M1 / 64, 256, 0, stream>>>(
      x, om1, WB2, proj_b, proj_gamma, proj_beta, proj_mean, proj_var,
      hbuf, H, W, 128, 13, 7);

  // ---- bilinear x2 upsample + residual ----
  upsample_add_kernel<<<8192, 256, 0, stream>>>(hbuf, pre_x, up_w, zbuf,
                                                B, O, H, W);

  // ---- node block (HW2 = 32768 -> shift 15, W2 = 256 -> shift 8) ----
  const int M2 = B * HW2;                             // 65536 pixels
  dcn_gemm_kernel<false, 2, 4, 4><<<M2 / 64, 256, 0, stream>>>(
      zbuf, nullptr, WB3, node_b_om, nullptr, nullptr, nullptr, nullptr,
      om2, H2, W2, 27, 15, 8);
  dcn_gemm_kernel<true, 8, 4, 4><<<M2 / 64, 256, 0, stream>>>(
      zbuf, om2, WB4, node_b, node_gamma, node_beta, node_mean, node_var,
      (float*)d_out, H2, W2, 128, 15, 8);
}
